// GAT_80668075753485
// MI455X (gfx1250) — compile-verified, matrix operand order checked by reading, not run on hardware
//
#include <hip/hip_runtime.h>
#include <hip/hip_bf16.h>
#include <math.h>

typedef __attribute__((ext_vector_type(16))) _Float16 v16h;
typedef __attribute__((ext_vector_type(8)))  float    v8f;

#define HEADS 8
#define HDIM  32
#define FOUT  256
#define NEG_SLOPE 0.2f

// ---------------------------------------------------------------------------
// Utility kernels
// ---------------------------------------------------------------------------
__global__ void zero_i32(int* __restrict__ p, int n) {
    int i = blockIdx.x * blockDim.x + threadIdx.x;
    if (i < n) p[i] = 0;
}

__global__ void count_deg(const int* __restrict__ dst, int* __restrict__ deg, int e) {
    int i = blockIdx.x * blockDim.x + threadIdx.x;
    if (i < e) atomicAdd(&deg[dst[i]], 1);
}

// Exclusive scan of deg[0..n) -> off[0..n], single 1024-thread block.
__global__ void __launch_bounds__(1024) scan_off(const int* __restrict__ deg,
                                                 int* __restrict__ off, int n) {
    __shared__ int sh[1024];
    int tid = threadIdx.x;
    int chunk = (n + 1023) >> 10;
    int s0 = tid * chunk;
    int s1 = s0 + chunk; if (s1 > n) s1 = n;
    int local = 0;
    for (int i = s0; i < s1; ++i) local += deg[i];
    sh[tid] = local;
    __syncthreads();
    for (int o = 1; o < 1024; o <<= 1) {
        int v = (tid >= o) ? sh[tid - o] : 0;
        __syncthreads();
        sh[tid] += v;
        __syncthreads();
    }
    int run = sh[tid] - local;      // exclusive base for this thread's chunk
    for (int i = s0; i < s1; ++i) { off[i] = run; run += deg[i]; }
    if (tid == 1023) off[n] = sh[1023];
}

// For each edge, place its SOURCE node id into the dst-grouped CSR list.
__global__ void scatter_edges(const int* __restrict__ src, const int* __restrict__ dst,
                              const int* __restrict__ off, int* __restrict__ cursor,
                              int* __restrict__ eidx, int e) {
    int i = blockIdx.x * blockDim.x + threadIdx.x;
    if (i < e) {
        int d = dst[i];
        int pos = atomicAdd(&cursor[d], 1);
        eidx[off[d] + pos] = src[i];
    }
}

__global__ void cvt_f32_to_f16(const float* __restrict__ in, _Float16* __restrict__ out, int n) {
    int i = blockIdx.x * blockDim.x + threadIdx.x;
    if (i < n) out[i] = (_Float16)in[i];
}

// W [K, 256] f32 row-major  ->  Wt [256, K] f16 row-major (column-major B for WMMA)
__global__ void transpose_w(const float* __restrict__ W, _Float16* __restrict__ Wt, int K) {
    int i = blockIdx.x * blockDim.x + threadIdx.x;   // i over K*256
    if (i < K * FOUT) {
        int k = i / FOUT, c = i % FOUT;
        Wt[c * K + k] = (_Float16)W[k * FOUT + c];
    }
}

// ---------------------------------------------------------------------------
// WMMA GEMM: C[M,256] = A[M,K](f16) * B[K,256] with Bt=[256,K] f16, f32 accum.
// One wave computes a 16x64 tile (4 accumulators, A reused across N-subtiles).
// ---------------------------------------------------------------------------
__device__ __forceinline__ v16h load_frag(const _Float16* __restrict__ p) {
    // elems 0..7 = K {0..7}, elems 8..15 = K {16..23} relative to lane's k-half
    v16h r;
#pragma unroll
    for (int e = 0; e < 8; ++e) { r[e] = p[e]; r[e + 8] = p[e + 16]; }
    return r;
}

__global__ void __launch_bounds__(256) gemm_wmma_f16(const _Float16* __restrict__ A,
                                                     const _Float16* __restrict__ Bt,
                                                     float* __restrict__ C,
                                                     int M, int K) {
    int gt   = blockIdx.x * 256 + threadIdx.x;
    int wid  = gt >> 5;
    int lane = gt & 31;
    int nWaves = (M >> 4) * (FOUT >> 6);
    if (wid >= nWaves) return;                  // wave-uniform

    int m0 = (wid >> 2) << 4;                   // 16-row block
    int n0 = (wid & 3) << 6;                    // 64-col block
    int l15 = lane & 15;
    int kh  = (lane >> 4) << 3;                 // k-half: 0 or 8

    const _Float16* pa  = A  + (size_t)(m0 + l15) * K + kh;
    const _Float16* pb0 = Bt + (size_t)(n0 + l15 +  0) * K + kh;
    const _Float16* pb1 = Bt + (size_t)(n0 + l15 + 16) * K + kh;
    const _Float16* pb2 = Bt + (size_t)(n0 + l15 + 32) * K + kh;
    const _Float16* pb3 = Bt + (size_t)(n0 + l15 + 48) * K + kh;

    v8f c0 = {}, c1 = {}, c2 = {}, c3 = {};
    for (int k0 = 0; k0 < K; k0 += 32) {
        v16h a  = load_frag(pa  + k0);
        v16h b0 = load_frag(pb0 + k0);
        v16h b1 = load_frag(pb1 + k0);
        v16h b2 = load_frag(pb2 + k0);
        v16h b3 = load_frag(pb3 + k0);
        c0 = __builtin_amdgcn_wmma_f32_16x16x32_f16(false, a, false, b0, (short)0, c0, false, false);
        c1 = __builtin_amdgcn_wmma_f32_16x16x32_f16(false, a, false, b1, (short)0, c1, false, false);
        c2 = __builtin_amdgcn_wmma_f32_16x16x32_f16(false, a, false, b2, (short)0, c2, false, false);
        c3 = __builtin_amdgcn_wmma_f32_16x16x32_f16(false, a, false, b3, (short)0, c3, false, false);
    }
    int rbase = (lane >> 4) << 3;               // rows 0..7 or 8..15
#pragma unroll
    for (int r = 0; r < 8; ++r) {
        size_t o = (size_t)(m0 + r + rbase) * FOUT + n0 + l15;
        C[o +  0] = c0[r];
        C[o + 16] = c1[r];
        C[o + 32] = c2[r];
        C[o + 48] = c3[r];
    }
}

// ---------------------------------------------------------------------------
// el/er: one thread per (node, head): dot(feat[n,h,:], al/ar[h,:])
// ---------------------------------------------------------------------------
__global__ void eler_kernel(const float* __restrict__ feat,
                            const float* __restrict__ al, const float* __restrict__ ar,
                            float* __restrict__ el, float* __restrict__ er, int n) {
    int i = blockIdx.x * blockDim.x + threadIdx.x;
    if (i >= n * HEADS) return;
    int node = i >> 3, hd = i & 7;
    const float* f = feat + (size_t)node * FOUT + hd * HDIM;
    const float* a = al + hd * HDIM;
    const float* b = ar + hd * HDIM;
    float sl = 0.f, sr = 0.f;
#pragma unroll
    for (int d = 0; d < HDIM; ++d) { float v = f[d]; sl += v * a[d]; sr += v * b[d]; }
    el[i] = sl; er[i] = sr;
}

// ---------------------------------------------------------------------------
// Fused edge-softmax + aggregation, pull mode: one wave32 per destination node.
// lane = head*4 + j. Three passes over incoming edges (max / sum-exp / accum).
// mode 0: out = elu(agg + b)   mode 1: out = agg + b + out (residual)
// ---------------------------------------------------------------------------
__global__ void __launch_bounds__(256) gat_pull(const float* __restrict__ feat,
                                                const float* __restrict__ el,
                                                const float* __restrict__ er,
                                                const int* __restrict__ off,
                                                const int* __restrict__ eidx,
                                                const float* __restrict__ bias,
                                                float* __restrict__ out,
                                                int n, int mode) {
    int gt   = blockIdx.x * 256 + threadIdx.x;
    int v    = gt >> 5;                          // node (wave id)
    int lane = gt & 31;
    if (v >= n) return;
    int hd = lane >> 2;                          // head 0..7
    int j  = lane & 3;                           // quarter within head group

    int beg = off[v], end = off[v + 1];
    float ern = er[(size_t)v * HEADS + hd];

    // Pass 1: per-head max of LeakyReLU(el[src]+er[v])
    float m = -INFINITY;
    for (int t = beg + j; t < end; t += 4) {
        int s = eidx[t];
        float e = el[(size_t)s * HEADS + hd] + ern;
        e = (e > 0.f) ? e : NEG_SLOPE * e;
        m = fmaxf(m, e);
    }
    m = fmaxf(m, __shfl_xor(m, 1));
    m = fmaxf(m, __shfl_xor(m, 2));

    // Pass 2: per-head sum of exp(e - max)
    float sum = 0.f;
    for (int t = beg + j; t < end; t += 4) {
        int s = eidx[t];
        float e = el[(size_t)s * HEADS + hd] + ern;
        e = (e > 0.f) ? e : NEG_SLOPE * e;
        sum += __expf(e - m);
    }
    sum += __shfl_xor(sum, 1);
    sum += __shfl_xor(sum, 2);
    float inv = (end > beg) ? 1.f / sum : 0.f;

    // Pass 3: all 32 lanes cooperatively accumulate one feat row per edge.
    // lane covers dims [j*8, j*8+8) of its head; feat row is L2-resident.
    float acc[8];
#pragma unroll
    for (int q = 0; q < 8; ++q) acc[q] = 0.f;
    for (int t = beg; t < end; ++t) {
        int s = eidx[t];
        if (t + 1 < end) {
            int sn = eidx[t + 1];
            __builtin_prefetch(feat + (size_t)sn * FOUT + hd * HDIM + j * 8, 0, 0);
        }
        float e = el[(size_t)s * HEADS + hd] + ern;
        e = (e > 0.f) ? e : NEG_SLOPE * e;
        float alpha = __expf(e - m) * inv;
        const float* fs = feat + (size_t)s * FOUT + hd * HDIM + j * 8;
#pragma unroll
        for (int q = 0; q < 8; ++q) acc[q] += fs[q] * alpha;
    }

    size_t base = (size_t)v * FOUT + hd * HDIM + j * 8;
#pragma unroll
    for (int q = 0; q < 8; ++q) {
        float val = acc[q] + bias[hd * HDIM + j * 8 + q];
        if (mode == 0) {
            val = (val > 0.f) ? val : (__expf(val) - 1.f);   // ELU
        } else {
            val += out[base + q];                            // residual
        }
        out[base + q] = val;
    }
}

// ---------------------------------------------------------------------------
// Host-side launch
// ---------------------------------------------------------------------------
static inline size_t align256(size_t x) { return (x + 255) & ~(size_t)255; }

extern "C" void kernel_launch(void* const* d_in, const int* in_sizes, int n_in,
                              void* d_out, int out_size, void* d_ws, size_t ws_size,
                              hipStream_t stream) {
    const int N = in_sizes[0] / 128;          // 50000
    const int E = in_sizes[1];                // 800000

    const float* x   = (const float*)d_in[0];
    const int*   src = (const int*)d_in[1];
    const int*   dst = (const int*)d_in[2];

    // Workspace carve-out
    char* ws = (char*)d_ws;
    size_t o = 0;
    int* off    = (int*)(ws + o); o = align256(o + (size_t)(N + 1) * 4);
    int* deg    = (int*)(ws + o); o = align256(o + (size_t)N * 4);
    int* cursor = (int*)(ws + o); o = align256(o + (size_t)N * 4);
    int* eidx   = (int*)(ws + o); o = align256(o + (size_t)E * 4);
    _Float16* h16  = (_Float16*)(ws + o); o = align256(o + (size_t)N * FOUT * 2);
    _Float16* wt16 = (_Float16*)(ws + o); o = align256(o + (size_t)FOUT * FOUT * 2);
    float* feat = (float*)(ws + o); o = align256(o + (size_t)N * FOUT * 4);
    float* el   = (float*)(ws + o); o = align256(o + (size_t)N * HEADS * 4);
    float* er   = (float*)(ws + o); o = align256(o + (size_t)N * HEADS * 4);
    (void)ws_size;

    float* h = (float*)d_out;                 // node features live in d_out

    const int T = 256;
    auto blk = [](int n, int t) { return (n + t - 1) / t; };

    // --- Build CSR (dst -> list of src), reused by all 4 layers -----------
    zero_i32<<<blk(N, T), T, 0, stream>>>(deg, N);
    zero_i32<<<blk(N, T), T, 0, stream>>>(cursor, N);
    count_deg<<<blk(E, T), T, 0, stream>>>(dst, deg, E);
    scan_off<<<1, 1024, 0, stream>>>(deg, off, N);
    scatter_edges<<<blk(E, T), T, 0, stream>>>(src, dst, off, cursor, eidx, E);

    // --- 4 GAT layers ------------------------------------------------------
    for (int layer = 0; layer < 4; ++layer) {
        const float* W  = (const float*)d_in[3 + 4 * layer];
        const float* al = (const float*)d_in[4 + 4 * layer];
        const float* ar = (const float*)d_in[5 + 4 * layer];
        const float* b  = (const float*)d_in[6 + 4 * layer];
        const int K = in_sizes[3 + 4 * layer] / FOUT;      // 128 or 256
        const float* hin = (layer == 0) ? x : h;

        // f16 staging for WMMA
        cvt_f32_to_f16<<<blk(N * K, T), T, 0, stream>>>(hin, h16, N * K);
        transpose_w<<<blk(K * FOUT, T), T, 0, stream>>>(W, wt16, K);

        // feat = h @ W  via v_wmma_f32_16x16x32_f16
        int nWaves = (N / 16) * (FOUT / 64);
        gemm_wmma_f16<<<blk(nWaves * 32, T), T, 0, stream>>>(h16, wt16, feat, N, K);

        // attention logits per node/head
        eler_kernel<<<blk(N * HEADS, T), T, 0, stream>>>(feat, al, ar, el, er, N);

        // fused softmax + aggregation + bias + activation/residual
        int mode = (layer == 0) ? 0 : 1;
        gat_pull<<<blk(N * 32, T), T, 0, stream>>>(feat, el, er, off, eidx, b, h, N, mode);
    }
}
